// Network_1726576855363
// MI455X (gfx1250) — matrix operand order repaired and checked by reading, not compiled, not run on hardware
//
#include <hip/hip_runtime.h>
#include <hip/hip_fp16.h>

typedef __attribute__((ext_vector_type(16))) _Float16 v16h;
typedef __attribute__((ext_vector_type(8)))  _Float16 v8h;
typedef __attribute__((ext_vector_type(8)))  float    v8f;
typedef __attribute__((ext_vector_type(4)))  unsigned int u32x4;
typedef __attribute__((ext_vector_type(8)))  int          i32x8;
typedef __attribute__((ext_vector_type(4)))  int          i32x4;

#if defined(__AMDGCN__) && __has_builtin(__builtin_amdgcn_tensor_load_to_lds) && __has_builtin(__builtin_amdgcn_s_wait_tensorcnt)
#define USE_TDM 1
#else
#define USE_TDM 0
#endif

__device__ __forceinline__ float lrelu_f(float v) { return v >= 0.0f ? v : 0.1f * v; }

// ---------------------------------------------------------------- elementwise
__global__ void k_f32_to_f16(const float* __restrict__ in, _Float16* __restrict__ out, int n) {
  int i = blockIdx.x * 256 + threadIdx.x;
  if (i < n) out[i] = (_Float16)in[i];
}

// Layer-1 weights: fp32 OIHW -> f16 [O][KdimP], k = i*R*S + r*S + s, zero-padded
__global__ void k_wt(const float* __restrict__ w, _Float16* __restrict__ wt,
                     int O, int Kdim, int KdimP) {
  int i = blockIdx.x * 256 + threadIdx.x;
  if (i >= O * KdimP) return;
  int k = i % KdimP, o = i / KdimP;
  wt[i] = (k < Kdim) ? (_Float16)w[o * Kdim + k] : (_Float16)0.0f;
}

// Layers 2-10 weights: fp32 OIHW -> f16 [O][Kdim], k = (r*S+s)*C + i  (channel-innermost)
__global__ void k_wt_nhwc(const float* __restrict__ w, _Float16* __restrict__ wt,
                          int O, int C, int RS) {
  int Kdim = C * RS;
  int i = blockIdx.x * 256 + threadIdx.x;
  if (i >= O * Kdim) return;
  int k = i % Kdim, o = i / Kdim;
  int ci = k % C, cell = k / C;
  wt[i] = (_Float16)w[((size_t)o * C + ci) * RS + cell];
}

// ---------------------------------------------------------------- TDM panel stage
// Copy 'halfs' f16 values (linear) from global 'src' into LDS offset 'ldsOff'.
__device__ __forceinline__ void tdm_stage_linear(unsigned ldsOff, const _Float16* src, unsigned halfs) {
#if USE_TDM
  unsigned long long ga = (unsigned long long)(uintptr_t)src;
  u32x4 g0;
  g0.x = 1u;                                           // count=1, user mode, no gather
  g0.y = ldsOff;                                       // LDS byte address
  g0.z = (unsigned)(ga & 0xffffffffull);               // global_addr[31:0]
  g0.w = (unsigned)((ga >> 32) & 0x1ffffffull) | (2u << 30);  // global_addr[56:32] | type=2
  i32x8 g1;
  g1[0] = (int)(1u << 16);                             // data_size=1 (2 bytes)
  g1[1] = (int)((halfs & 0xffffu) << 16);              // tensor_dim0[15:0]
  g1[2] = (int)(((halfs >> 16) & 0xffffu) | (1u << 16)); // tensor_dim0[31:16] | tensor_dim1=1
  g1[3] = (int)((halfs & 0xffffu) << 16);              // tile_dim0 = halfs (1-D tile)
  g1[4] = 0;                                           // tile_dim1=0, tile_dim2=0
  g1[5] = (int)halfs;                                  // tensor_dim0_stride[31:0]
  g1[6] = 0;
  g1[7] = 0;
  i32x4 gz = {};
#if __clang_major__ >= 23
  i32x8 gz8 = {};
  __builtin_amdgcn_tensor_load_to_lds(g0, g1, gz, gz, gz8, 0);
#else
  __builtin_amdgcn_tensor_load_to_lds(g0, g1, gz, gz, 0);
#endif
  __builtin_amdgcn_s_wait_tensorcnt(0);
#else
  (void)ldsOff; (void)src; (void)halfs;
#endif
}

// ---------------------------------------------------------------- 3x3 conv, NHWC, WMMA
// Block = 8 waves sharing one 32-output-channel group (B panel staged in LDS);
// each wave computes a 16-pixel x 32-channel tile via 2 WMMAs per 32-wide K step.
template<int C>
__global__ __launch_bounds__(256)
void k_conv3(const _Float16* __restrict__ in,   // [B][H][W][C]   f16
             const _Float16* __restrict__ wt,   // [O][Kdim]      f16, k=(ry*3+sx)*C+ci
             const float*    __restrict__ bias, // [O]
             _Float16*       __restrict__ out,  // [B][Ho][Wo][O] f16
             int Bn, int H, int W, int O, int Ho, int Wo, int stride)
{
  constexpr int Kdim = 9 * C;
  extern __shared__ __align__(16) _Float16 sB[];

  const int lane = threadIdx.x & 31;
  const int wave = threadIdx.x >> 5;
  const int mTiles  = (Bn * Ho * Wo) >> 4;
  const int mGroups = mTiles >> 3;
  const int nt = blockIdx.x / mGroups;
  const int mg = blockIdx.x - nt * mGroups;
  const int mt = (mg << 3) + wave;

  // ---- stage B panel: 32 rows x Kdim halfs into LDS (one DMA per block)
  const _Float16* wpanel = wt + (size_t)(nt << 5) * Kdim;
#if USE_TDM
  if (threadIdx.x < 32) {
    unsigned ldsOff = (unsigned)(uintptr_t)(void*)sB;
    tdm_stage_linear(ldsOff, wpanel, 32u * (unsigned)Kdim);
  }
#else
  for (int i = threadIdx.x; i < (32 * Kdim) / 8; i += 256)
    ((v8h*)sB)[i] = ((const v8h*)wpanel)[i];
#endif
  __syncthreads();

  // ---- per-lane A row: pixel m = mt*16 + (lane&15)
  const int m = (mt << 4) + (lane & 15);
  const int HoWo = Ho * Wo;
  const int b   = m / HoWo;
  const int rem = m - b * HoWo;
  const int oy  = rem / Wo;
  const int ox  = rem - oy * Wo;
  const int iy0 = oy * stride - 1;
  const int ix0 = ox * stride - 1;
  const int half = lane >> 4;
  const size_t inBase = (size_t)b * H * W * C;

  v8f acc0 = {};
  v8f acc1 = {};
  const _Float16* b0p = sB + (size_t)(lane & 15) * Kdim + (half << 3);
  const _Float16* b1p = b0p + (size_t)16 * Kdim;

  for (int kb = 0; kb < Kdim; kb += 32) {
    // K step lies inside a single (ry,sx) cell since C % 32 == 0
    const int cell = kb / C;
    const int ci0  = kb - cell * C;
    const int ry   = cell / 3;
    const int sx   = cell - ry * 3;
    const int iy   = iy0 + ry;
    const int ix   = ix0 + sx;
    const bool ok  = (iy >= 0) && (iy < H) && (ix >= 0) && (ix < W);
    const int iyc  = iy < 0 ? 0 : (iy >= H ? H - 1 : iy);
    const int ixc  = ix < 0 ? 0 : (ix >= W ? W - 1 : ix);

    const _Float16* ap = in + inBase + ((size_t)iyc * W + ixc) * C + ci0 + (half << 3);
    v8h z = {};
    v8h a0 = *(const v8h*)ap;          // K-local: half*8 + 0..7
    v8h a1 = *(const v8h*)(ap + 16);   // K-local: 16 + half*8 + 0..7
    a0 = ok ? a0 : z;
    a1 = ok ? a1 : z;
    v16h a = __builtin_shufflevector(a0, a1, 0,1,2,3,4,5,6,7,8,9,10,11,12,13,14,15);

    v8h w0l = *(const v8h*)(b0p + kb), w0h = *(const v8h*)(b0p + kb + 16);
    v8h w1l = *(const v8h*)(b1p + kb), w1h = *(const v8h*)(b1p + kb + 16);
    v16h bw0 = __builtin_shufflevector(w0l, w0h, 0,1,2,3,4,5,6,7,8,9,10,11,12,13,14,15);
    v16h bw1 = __builtin_shufflevector(w1l, w1h, 0,1,2,3,4,5,6,7,8,9,10,11,12,13,14,15);

    acc0 = __builtin_amdgcn_wmma_f32_16x16x32_f16(false, a, false, bw0, (short)0, acc0, false, false);
    acc1 = __builtin_amdgcn_wmma_f32_16x16x32_f16(false, a, false, bw1, (short)0, acc1, false, false);
  }

  // ---- epilogue: NHWC stores, contiguous across lanes
  const int n0 = (nt << 5) + (lane & 15);
  const int n1 = n0 + 16;
  const float bv0 = bias[n0];
  const float bv1 = bias[n1];
  #pragma unroll
  for (int r = 0; r < 8; ++r) {
    const size_t mm = (size_t)((mt << 4) + r + (half << 3));
    out[mm * O + n0] = (_Float16)lrelu_f(acc0[r] + bv0);
    out[mm * O + n1] = (_Float16)lrelu_f(acc1[r] + bv1);
  }
}

// ---------------------------------------------------------------- 7x7 conv (layer 1)
// NCHW f16 input (C=3), gathered im2col with clamp+select (no EXEC divergence),
// NHWC f16 output. O=32 -> single 32-channel group.
__global__ __launch_bounds__(256)
void k_conv7(const _Float16* __restrict__ in,   // [B][3][H][W] f16
             const _Float16* __restrict__ wt,   // [32][160]    f16, k=i*49+r*7+s
             const float*    __restrict__ bias, // [32]
             _Float16*       __restrict__ out,  // [B][H][W][32] f16
             int Bn, int C, int H, int W, int O)
{
  constexpr int RS = 49, S = 7, Kdim = 147, KdimP = 160;
  const int lane = threadIdx.x & 31;
  const int wave = threadIdx.x >> 5;
  const int mt = blockIdx.x * 8 + wave;

  const int m  = (mt << 4) + (lane & 15);
  const int HW = H * W;
  const int b   = m / HW;
  const int rem = m - b * HW;
  const int oy  = rem / W;
  const int ox  = rem - oy * W;
  const int iy0 = oy - 3;
  const int ix0 = ox - 3;
  const int half = lane >> 4;
  const size_t inBase = (size_t)b * C * HW;

  v8f acc0 = {};
  v8f acc1 = {};
  const _Float16* wrow0 = wt + (size_t)(lane & 15) * KdimP + (half << 4);
  const _Float16* wrow1 = wrow0 + (size_t)16 * KdimP;

  for (int kb = 0; kb < KdimP; kb += 32) {
    v16h a;
    #pragma unroll
    for (int e = 0; e < 16; ++e) {
      const int k = kb + (half << 4) + e;
      bool ok = k < Kdim;
      const int kk = ok ? k : 0;
      const int ci = kk / RS;
      const int rs = kk - ci * RS;
      const int ry = rs / S;
      const int sx = rs - ry * S;
      const int iy = iy0 + ry;
      const int ix = ix0 + sx;
      ok = ok && (iy >= 0) && (iy < H) && (ix >= 0) && (ix < W);
      const int iyc = iy < 0 ? 0 : (iy >= H ? H - 1 : iy);
      const int ixc = ix < 0 ? 0 : (ix >= W ? W - 1 : ix);
      const _Float16 v = in[inBase + ((size_t)ci * H + iyc) * W + ixc];
      a[e] = ok ? v : (_Float16)0.0f;
    }
    v16h b0, b1;
    #pragma unroll
    for (int e = 0; e < 16; ++e) { b0[e] = wrow0[kb + e]; b1[e] = wrow1[kb + e]; }

    acc0 = __builtin_amdgcn_wmma_f32_16x16x32_f16(false, a, false, b0, (short)0, acc0, false, false);
    acc1 = __builtin_amdgcn_wmma_f32_16x16x32_f16(false, a, false, b1, (short)0, acc1, false, false);
  }

  const int n0 = lane & 15;
  const int n1 = n0 + 16;
  const float bv0 = bias[n0];
  const float bv1 = bias[n1];
  #pragma unroll
  for (int r = 0; r < 8; ++r) {
    const size_t mm = (size_t)((mt << 4) + r + (half << 3));
    out[mm * O + n0] = (_Float16)lrelu_f(acc0[r] + bv0);
    out[mm * O + n1] = (_Float16)lrelu_f(acc1[r] + bv1);
  }
}

// ---------------------------------------------------------------- bilinear warp (NHWC)
__global__ void k_warp(const _Float16* __restrict__ f2, const float* __restrict__ flow,
                       _Float16* __restrict__ out, int Bn, int C, int H, int W, float scale)
{
  int idx = blockIdx.x * 256 + threadIdx.x;
  int total = Bn * H * W * C;
  if (idx >= total) return;
  int c = idx % C; int t = idx / C;
  int x = t % W;  t /= W;
  int y = t % H;  int b = t / H;

  float fx = flow[(((size_t)b * 2 + 0) * H + y) * W + x] * scale;
  float fy = flow[(((size_t)b * 2 + 1) * H + y) * W + x] * scale;
  float px = (float)x + fx;
  float py = (float)y + fy;
  float x0 = floorf(px), y0 = floorf(py);
  float wx = px - x0,   wy = py - y0;
  int x0i = (int)x0, y0i = (int)y0;

  const _Float16* img = f2 + (size_t)b * H * W * C + c;
  auto g = [&](int yi, int xi) -> float {
    bool valid = (xi >= 0) && (xi <= W - 1) && (yi >= 0) && (yi <= H - 1);
    int xc = xi < 0 ? 0 : (xi > W - 1 ? W - 1 : xi);
    int yc = yi < 0 ? 0 : (yi > H - 1 ? H - 1 : yi);
    float v = (float)img[((size_t)yc * W + xc) * C];
    return valid ? v : 0.0f;
  };
  float o = g(y0i, x0i)         * (1.0f - wx) * (1.0f - wy)
          + g(y0i, x0i + 1)     * wx          * (1.0f - wy)
          + g(y0i + 1, x0i)     * (1.0f - wx) * wy
          + g(y0i + 1, x0i + 1) * wx          * wy;
  out[idx] = (_Float16)o;
}

// ---------------------------------------------------------------- correlation (NHWC in, NCHW f32 out)
__global__ void k_corr(const _Float16* __restrict__ f1, const _Float16* __restrict__ warped,
                       float* __restrict__ out, int Bn, int C, int H, int W)
{
  int idx = blockIdx.x * 256 + threadIdx.x;
  int total = Bn * 49 * H * W;
  if (idx >= total) return;
  int x = idx % W; int t = idx / W;
  int y = t % H;  t /= H;
  int n = t % 49; int b = t / 49;
  int i = n / 7, j = n % 7;
  int sy = y + i - 3;
  int sx = x + j - 3;

  float acc = 0.0f;
  if (sy >= 0 && sy < H && sx >= 0 && sx < W) {
    const _Float16* a = f1     + (((size_t)b * H + y)  * W + x)  * C;
    const _Float16* s = warped + (((size_t)b * H + sy) * W + sx) * C;
    for (int c = 0; c < C; ++c)
      acc += (float)a[c] * (float)s[c];
  }
  acc /= (float)C;
  out[idx] = lrelu_f(acc);
}

// ---------------------------------------------------------------- host
struct LayerCfg { int C, H, W, O, Ho, Wo, R, stride, pad, KdimP; };
static const LayerCfg LC[10] = {
  {   3, 384, 768,  32, 384, 768, 7, 1, 3,  160 },
  {  32, 384, 768,  32, 192, 384, 3, 2, 1,  288 },
  {  32, 192, 384,  32, 192, 384, 3, 1, 1,  288 },
  {  32, 192, 384,  32, 192, 384, 3, 1, 1,  288 },
  {  32, 192, 384,  64,  96, 192, 3, 2, 1,  288 },
  {  64,  96, 192,  64,  96, 192, 3, 1, 1,  576 },
  {  64,  96, 192,  96,  48,  96, 3, 2, 1,  576 },
  {  96,  48,  96,  96,  48,  96, 3, 1, 1,  864 },
  {  96,  48,  96, 128,  24,  48, 3, 2, 1,  864 },
  { 128,  24,  48, 192,  12,  24, 3, 2, 1, 1152 },
};

static void launch_conv3(int C, int blocks, size_t smem, hipStream_t stream,
                         const _Float16* in, const _Float16* wt, const float* bias,
                         _Float16* out, int Bn, int H, int W, int O, int Ho, int Wo, int stride) {
  switch (C) {
    case 32:  k_conv3<32> <<<blocks, 256, smem, stream>>>(in, wt, bias, out, Bn, H, W, O, Ho, Wo, stride); break;
    case 64:  k_conv3<64> <<<blocks, 256, smem, stream>>>(in, wt, bias, out, Bn, H, W, O, Ho, Wo, stride); break;
    case 96:  k_conv3<96> <<<blocks, 256, smem, stream>>>(in, wt, bias, out, Bn, H, W, O, Ho, Wo, stride); break;
    default:  k_conv3<128><<<blocks, 256, smem, stream>>>(in, wt, bias, out, Bn, H, W, O, Ho, Wo, stride); break;
  }
}

extern "C" void kernel_launch(void* const* d_in, const int* in_sizes, int n_in,
                              void* d_out, int out_size, void* d_ws, size_t ws_size,
                              hipStream_t stream) {
  const int B = 4;
  const float* img[2] = { (const float*)d_in[0], (const float*)d_in[1] };
  const float* flow   = (const float*)d_in[2];

  char* ws = (char*)d_ws;
  size_t off = 0;
  auto alloc = [&](size_t bytes) -> char* {
    char* p = ws + off;
    off = (off + bytes + 255) & ~(size_t)255;
    return p;
  };

  _Float16* wt[10];
  for (int i = 0; i < 10; ++i)
    wt[i] = (_Float16*)alloc((size_t)LC[i].O * LC[i].KdimP * sizeof(_Float16));

  const size_t maxAct = (size_t)B * 32 * 384 * 768;
  _Float16* actA = (_Float16*)alloc(maxAct * sizeof(_Float16));
  _Float16* actB = (_Float16*)alloc(maxAct * sizeof(_Float16));

  const size_t featElems = (size_t)B * 192 * 12 * 24;
  _Float16* feat[2] = { (_Float16*)alloc(featElems * sizeof(_Float16)),
                        (_Float16*)alloc(featElems * sizeof(_Float16)) };
  _Float16* warped  = (_Float16*)alloc(featElems * sizeof(_Float16));

  // 1) repack weights
  {
    int total = LC[0].O * LC[0].KdimP;
    k_wt<<<(total + 255) / 256, 256, 0, stream>>>((const float*)d_in[3], wt[0], LC[0].O, 147, LC[0].KdimP);
  }
  for (int i = 1; i < 10; ++i) {
    int total = LC[i].O * LC[i].KdimP;
    k_wt_nhwc<<<(total + 255) / 256, 256, 0, stream>>>(
        (const float*)d_in[3 + 2 * i], wt[i], LC[i].O, LC[i].C, 9);
  }

  // 2) feature pyramid for both images (ping-pong; only last level kept)
  for (int im = 0; im < 2; ++im) {
    int n0 = B * 3 * 384 * 768;
    k_f32_to_f16<<<(n0 + 255) / 256, 256, 0, stream>>>(img[im], actB, n0);
    const _Float16* cur = actB;
    for (int i = 0; i < 10; ++i) {
      const LayerCfg& L = LC[i];
      _Float16* dst = (i == 9) ? feat[im] : (((i & 1) == 0) ? actA : actB);
      int mTiles = (B * L.Ho * L.Wo) >> 4;
      const float* bias = (const float*)d_in[4 + 2 * i];
      if (i == 0) {
        k_conv7<<<mTiles / 8, 256, 0, stream>>>(cur, wt[0], bias, dst, B, 3, L.H, L.W, L.O);
      } else {
        int nTiles = L.O >> 5;
        int blocks = (mTiles / 8) * nTiles;
        size_t smem = (size_t)32 * L.KdimP * sizeof(_Float16);
        launch_conv3(L.C, blocks, smem, stream, cur, wt[i], bias, dst,
                     B, L.H, L.W, L.O, L.Ho, L.Wo, L.stride);
      }
      cur = dst;
    }
  }

  // 3) warp f2 by 0.625*flow, 4) correlation + lrelu -> d_out (fp32 NCHW)
  int nW = (int)featElems;
  k_warp<<<(nW + 255) / 256, 256, 0, stream>>>(feat[1], flow, warped, B, 192, 12, 24, 0.625f);
  int nC = B * 49 * 12 * 24;
  k_corr<<<(nC + 255) / 256, 256, 0, stream>>>(feat[0], warped, (float*)d_out, B, 192, 12, 24);
}
